// MaskedMultiheadAttention_36953898615180
// MI455X (gfx1250) — matrix operand order repaired and checked by reading, not compiled, not run on hardware
//
#include <hip/hip_runtime.h>
#include <hip/hip_bf16.h>

// ---------------------------------------------------------------------------
// Masked multi-head attention, MI455X (gfx1250, wave32).
//   Stage 0: one-shot fp32 -> bf16 conversion of X and Wq/Wk/Wv (elementwise,
//            bandwidth bound, off the matrix critical path).
//   Stage 1: QKV projection GEMMs, all-bf16 operands via global_load_b128,
//            v_wmma_f32_16x16x32_bf16, 32x64 register tile per wave.
//            V is stored TRANSPOSED [B,H,Dh,T] so stage 2's PV operand is
//            contiguous along the WMMA K dimension.
//   Stage 2: causal flash attention (online softmax); scores + PV via WMMA,
//            P re-layout through LDS (per-wave; DS ops are in-order per wave).
// Problem is compute bound (~50 GFLOP vs ~50 MB @ 23.3 TB/s): keep the WMMA
// pipe fed, keep VALU/branches out of the inner loops.
// ---------------------------------------------------------------------------

#define T_SEQ 2048
#define C_DIM 1024
#define NHEAD 16
#define DHEAD 64
#define BATCH 2
#define QKV_ELEMS ((size_t)BATCH * NHEAD * T_SEQ * DHEAD)   // 4M elems
#define W_ELEMS   ((size_t)C_DIM * C_DIM)                   // 1M elems

// workspace layout (bf16 / ushort elements)
#define QOFF ((size_t)0)
#define KOFF (QKV_ELEMS)
#define VOFF (2 * QKV_ELEMS)          // transposed [B,H,Dh,T]
#define XOFF (3 * QKV_ELEMS)
#define WOFF (4 * QKV_ELEMS)          // Wq,Wk,Wv bf16, concatenated

typedef __attribute__((ext_vector_type(16))) __bf16 v16bf;
typedef __attribute__((ext_vector_type(8)))  float  v8f;

union BFrag { unsigned u[8]; uint4 q[2]; v16bf v; };

static __device__ __forceinline__ unsigned short f2bf(float f) {
  unsigned x = __float_as_uint(f);
  x += 0x7FFFu + ((x >> 16) & 1u);      // round-to-nearest-even
  return (unsigned short)(x >> 16);
}
static __device__ __forceinline__ unsigned packbf(float lo, float hi) {
  return (unsigned)f2bf(lo) | ((unsigned)f2bf(hi) << 16);
}

static __device__ __forceinline__ v8f wmma_bf16(v16bf a, v16bf b, v8f c) {
  return __builtin_amdgcn_wmma_f32_16x16x32_bf16(false, a, false, b,
                                                 (short)0, c, false, false);
}

// ---------------------------------------------------------------------------
// Stage 0: fp32 -> bf16, 8 elements / thread (n must be a multiple of 8)
// ---------------------------------------------------------------------------
__global__ __launch_bounds__(256) void cvt_bf16_kernel(
    const float* __restrict__ src, unsigned short* __restrict__ dst, int n)
{
  const int i = (blockIdx.x * 256 + threadIdx.x) * 8;
  if (i >= n) return;
  float4 a = *(const float4*)(src + i);
  float4 b = *(const float4*)(src + i + 4);
  uint4 o;
  o.x = packbf(a.x, a.y);
  o.y = packbf(a.z, a.w);
  o.z = packbf(b.x, b.y);
  o.w = packbf(b.z, b.w);
  *(uint4*)(dst + i) = o;
}

// ---------------------------------------------------------------------------
// Stage 1: Q/K/V = X @ W.T + b.  One wave -> 32x64 output tile.
// A frag (16-bit 16x32, ISA 7.12.2): lane M = lane%16, half = lane/16,
//   u[0..3] = K half*8..half*8+7, u[4..7] = K 16+half*8.. -> 2 x b128.
// B frag (32x16): lane N = lane%16, same K order as A -> 2 x b128.
// ---------------------------------------------------------------------------
__global__ __launch_bounds__(256) void qkv_gemm_kernel(
    const unsigned short* __restrict__ ws_ro,
    const float* __restrict__ bq, const float* __restrict__ bk,
    const float* __restrict__ bv,
    unsigned short* __restrict__ ws)
{
  const int lane = threadIdx.x & 31;
  const int half = lane >> 4;
  const int nn   = lane & 15;
  const int wid  = blockIdx.x * 8 + (threadIdx.x >> 5);
  const int ng = wid & 15;            // 16 col groups of 64
  const int mg = (wid >> 4) & 127;    // 128 row groups of 32
  const int g  = wid >> 11;           // 0:Q 1:K 2:V

  const unsigned short* Xbf = ws_ro + XOFF;
  const unsigned short* Wm  = ws_ro + WOFF + (size_t)g * W_ELEMS;
  const float* bias = (g == 0) ? bq : (g == 1) ? bk : bv;

  const int row0 = mg * 32;
  const int col0 = ng * 64;

  v8f acc0[4], acc1[4];
  #pragma unroll
  for (int nt = 0; nt < 4; ++nt) { acc0[nt] = {}; acc1[nt] = {}; }

  const unsigned short* Ar0 = Xbf + (size_t)(row0 + nn) * C_DIM;
  const unsigned short* Ar1 = Ar0 + 16 * C_DIM;

  for (int k0 = 0; k0 < C_DIM; k0 += 32) {
    // Unconditional speculative prefetch of next K tile (stays inside the
    // workspace allocation; keeps the loop a single straight-line block).
    __builtin_prefetch(Ar0 + k0 + 32, 0, 1);
    __builtin_prefetch(Ar1 + k0 + 32, 0, 1);

    BFrag a0, a1;
    a0.q[0] = *(const uint4*)(Ar0 + k0 + half * 8);
    a0.q[1] = *(const uint4*)(Ar0 + k0 + 16 + half * 8);
    a1.q[0] = *(const uint4*)(Ar1 + k0 + half * 8);
    a1.q[1] = *(const uint4*)(Ar1 + k0 + 16 + half * 8);
    #pragma unroll
    for (int nt = 0; nt < 4; ++nt) {
      const unsigned short* Br =
          Wm + (size_t)(col0 + nt * 16 + nn) * C_DIM + k0 + half * 8;
      BFrag bf;
      bf.q[0] = *(const uint4*)(Br);
      bf.q[1] = *(const uint4*)(Br + 16);
      acc0[nt] = wmma_bf16(a0.v, bf.v, acc0[nt]);
      acc1[nt] = wmma_bf16(a1.v, bf.v, acc1[nt]);
    }
  }

  // bias + store bf16.  Q,K: [B,H,T,Dh].  V: transposed [B,H,Dh,T].
  #pragma unroll
  for (int nt = 0; nt < 4; ++nt) {
    const int col = col0 + nt * 16 + nn;
    const float bb = bias[col];
    const int h = col >> 6, d = col & 63;
    #pragma unroll
    for (int mi = 0; mi < 2; ++mi) {
      #pragma unroll
      for (int r = 0; r < 8; ++r) {
        const int row = row0 + mi * 16 + half * 8 + r;
        const int b   = row >> 11;             // /T_SEQ
        const int t   = row & (T_SEQ - 1);
        const float val = (mi ? acc1[nt][r] : acc0[nt][r]) + bb;
        const size_t bh = (size_t)(b * NHEAD + h);
        size_t idx;
        if (g == 2) idx = VOFF + (bh * DHEAD + d) * T_SEQ + t;      // V^T
        else        idx = (g ? KOFF : QOFF) + (bh * T_SEQ + t) * DHEAD + d;
        ws[idx] = f2bf(val);
      }
    }
  }
}

// ---------------------------------------------------------------------------
// Stage 2: causal flash attention. 4 waves/block, 16 query rows per wave,
// keys processed 32 at a time.
// ---------------------------------------------------------------------------
__global__ __launch_bounds__(128) void attn_kernel(
    const unsigned short* __restrict__ ws, float* __restrict__ out)
{
  __shared__ __align__(16) unsigned short sP[4][16][32];   // per-wave P tile

  const int lane = threadIdx.x & 31;
  const int half = lane >> 4;
  const int nn   = lane & 15;
  const int wave = threadIdx.x >> 5;
  const int bh   = blockIdx.x >> 5;        // B*H = 32
  const int qblk = blockIdx.x & 31;        // T/64 query blocks
  const int qt0  = qblk * 64 + wave * 16;
  const int b = bh >> 4, h = bh & 15;

  const unsigned short* Q  = ws + QOFF;
  const unsigned short* K  = ws + KOFF;
  const unsigned short* Vt = ws + VOFF;
  const size_t hb  = (size_t)bh * T_SEQ * DHEAD;   // Q/K base
  const size_t hbv = (size_t)bh * DHEAD * T_SEQ;   // V^T base

  // Q A-fragments for d in [0,32) and [32,64)
  BFrag aQ[2];
  {
    const unsigned short* qrow = Q + hb + (size_t)(qt0 + nn) * DHEAD;
    #pragma unroll
    for (int f = 0; f < 2; ++f) {
      aQ[f].q[0] = *(const uint4*)(qrow + f * 32 + half * 8);
      aQ[f].q[1] = *(const uint4*)(qrow + f * 32 + 16 + half * 8);
    }
  }

  float mrow[8], lrow[8];
  v8f acc[4];
  #pragma unroll
  for (int r = 0; r < 8; ++r) { mrow[r] = -__builtin_inff(); lrow[r] = 0.f; }
  #pragma unroll
  for (int dc = 0; dc < 4; ++dc) acc[dc] = {};

  const int q_hi = qt0 + 15;
  const float scale = 0.125f;              // 1/sqrt(64)

  for (int kc = 0; kc <= q_hi; kc += 32) {
    // ---- scores: 2 key sub-tiles x (Dh=64 -> two K=32 WMMA steps)
    v8f s[2];
    #pragma unroll
    for (int tg = 0; tg < 2; ++tg) {
      const unsigned short* krow = K + hb + (size_t)(kc + tg * 16 + nn) * DHEAD;
      v8f c = {};
      #pragma unroll
      for (int f = 0; f < 2; ++f) {
        BFrag bK;              // B[k=d][n=key], d contiguous in memory
        bK.q[0] = *(const uint4*)(krow + f * 32 + half * 8);
        bK.q[1] = *(const uint4*)(krow + f * 32 + 16 + half * 8);
        c = wmma_bf16(aQ[f].v, bK.v, c);
      }
      s[tg] = c;
    }

    // ---- scale + causal mask + chunk row-max (16-lane column groups)
    float cmax[8];
    #pragma unroll
    for (int r = 0; r < 8; ++r) {
      const int mg2 = qt0 + half * 8 + r;
      float s0 = s[0][r] * scale;
      float s1 = s[1][r] * scale;
      if (kc + nn      > mg2) s0 = -__builtin_inff();
      if (kc + 16 + nn > mg2) s1 = -__builtin_inff();
      s[0][r] = s0; s[1][r] = s1;
      float m = fmaxf(s0, s1);
      m = fmaxf(m, __shfl_xor(m, 1));
      m = fmaxf(m, __shfl_xor(m, 2));
      m = fmaxf(m, __shfl_xor(m, 4));
      m = fmaxf(m, __shfl_xor(m, 8));
      cmax[r] = m;
    }

    // ---- online softmax update, stage P (bf16) into LDS
    #pragma unroll
    for (int r = 0; r < 8; ++r) {
      const float mnew = fmaxf(mrow[r], cmax[r]);
      const float fac  = __expf(mrow[r] - mnew);
      const float p0   = __expf(s[0][r] - mnew);
      const float p1   = __expf(s[1][r] - mnew);
      mrow[r] = mnew;
      float cs = p0 + p1;
      cs += __shfl_xor(cs, 1);
      cs += __shfl_xor(cs, 2);
      cs += __shfl_xor(cs, 4);
      cs += __shfl_xor(cs, 8);
      lrow[r] = lrow[r] * fac + cs;
      const int m = half * 8 + r;
      sP[wave][m][nn]      = f2bf(p0);
      sP[wave][m][16 + nn] = f2bf(p1);
      #pragma unroll
      for (int dc = 0; dc < 4; ++dc) acc[dc][r] *= fac;
    }

    // ---- P as A-fragment (same-wave LDS; DS in-order => no barrier)
    BFrag aP;
    aP.q[0] = *(const uint4*)(&sP[wave][nn][half * 8]);
    aP.q[1] = *(const uint4*)(&sP[wave][nn][16 + half * 8]);

    // ---- P @ V : V^T is contiguous along keys -> 2 x b128 per fragment
    #pragma unroll
    for (int dc = 0; dc < 4; ++dc) {
      const unsigned short* vrow =
          Vt + hbv + (size_t)(dc * 16 + nn) * T_SEQ + kc + half * 8;
      BFrag bV;              // B[k=key][n=d], key pairs contiguous
      bV.q[0] = *(const uint4*)(vrow);
      bV.q[1] = *(const uint4*)(vrow + 16);
      acc[dc] = wmma_bf16(aP.v, bV.v, acc[dc]);
    }
  }

  // ---- normalize and store fp32 output [B,T,C]
  #pragma unroll
  for (int r = 0; r < 8; ++r) {
    const int t = qt0 + half * 8 + r;
    const float inv = 1.0f / lrow[r];
    #pragma unroll
    for (int dc = 0; dc < 4; ++dc)
      out[(size_t)(b * T_SEQ + t) * C_DIM + h * DHEAD + dc * 16 + nn] =
          acc[dc][r] * inv;
  }
}

// ---------------------------------------------------------------------------
extern "C" void kernel_launch(void* const* d_in, const int* in_sizes, int n_in,
                              void* d_out, int out_size, void* d_ws, size_t ws_size,
                              hipStream_t stream) {
  const float* X  = (const float*)d_in[0];
  const float* Wq = (const float*)d_in[1];
  const float* bq = (const float*)d_in[2];
  const float* Wk = (const float*)d_in[3];
  const float* bk = (const float*)d_in[4];
  const float* Wv = (const float*)d_in[5];
  const float* bv = (const float*)d_in[6];
  unsigned short* ws = (unsigned short*)d_ws;
  float* out = (float*)d_out;

  // Stage 0: one-shot bf16 conversion (X: 4M elems, each W: 1M elems)
  cvt_bf16_kernel<<<(int)(QKV_ELEMS / (8 * 256)), 256, 0, stream>>>(
      X, ws + XOFF, (int)QKV_ELEMS);
  cvt_bf16_kernel<<<(int)(W_ELEMS / (8 * 256)), 256, 0, stream>>>(
      Wq, ws + WOFF + 0 * W_ELEMS, (int)W_ELEMS);
  cvt_bf16_kernel<<<(int)(W_ELEMS / (8 * 256)), 256, 0, stream>>>(
      Wk, ws + WOFF + 1 * W_ELEMS, (int)W_ELEMS);
  cvt_bf16_kernel<<<(int)(W_ELEMS / (8 * 256)), 256, 0, stream>>>(
      Wv, ws + WOFF + 2 * W_ELEMS, (int)W_ELEMS);

  // Stage 1: 3 mats * 128 row-groups * 16 col-groups = 6144 waves / 8 = 768
  qkv_gemm_kernel<<<768, 256, 0, stream>>>(ws, bq, bk, bv, ws);

  // Stage 2: B*H * T/64 = 1024 blocks, 4 waves each
  attn_kernel<<<1024, 128, 0, stream>>>(ws, out);
}
